// XFeatLighterGlueONNX_68350109549125
// MI455X (gfx1250) — compile-verified, matrix operand order checked by reading, not run on hardware
//
#include <hip/hip_runtime.h>
#include <hip/hip_bf16.h>

typedef __attribute__((ext_vector_type(16))) _Float16 v16h;
typedef __attribute__((ext_vector_type(8)))  float    v8f;

#define HF 60
#define WF 80
#define SP (HF*WF)            // 4800
#define HI 480
#define WI 640
#define NPIX (HI*WI)          // 307200
#define KSEL 4096
#define NCH 64

// ---- output layout (floats) ----
#define O_KPTS0   0
#define O_KPTS1   8192
#define O_MATCH01 16384
#define O_SCORE0  20480
#define O_VALID   24576
#define O_H0      28672
#define O_H1      (28672+307200)
#define O_R0      (28672+2*307200)
#define O_R1      (28672+3*307200)

// ---- workspace layout (bytes) ----
#define WS_NFEATS  0u                     // 2*64*4800 f32 = 2457600
#define WS_HEAT    2457600u               // 2*307200 f32  = 2457600
#define WS_SEL     4915200u               // 2*4096 u64    = 65536
#define WS_DESC    4980736u               // 2*4096*64 f16 = 1048576
#define WS_ROWMAX  6029312u               // 4096 u64
#define WS_COLMAX  6062080u               // 4096 u64
#define WS_HIST    6094848u               // 256 u32
#define WS_CTL     6095872u               // ctl block

struct Ctl {
  unsigned long long prefix;
  unsigned long long mask;
  unsigned int rank;
  unsigned int cnt;
};

__device__ __forceinline__ unsigned fordf(float f) {
  unsigned u = __float_as_uint(f);
  return (u & 0x80000000u) ? ~u : (u | 0x80000000u);
}
__device__ __forceinline__ float finvord(unsigned o) {
  unsigned u = (o & 0x80000000u) ? (o ^ 0x80000000u) : ~o;
  return __uint_as_float(u);
}
__device__ __forceinline__ unsigned long long shflxor_u64(unsigned long long v, int m) {
  unsigned lo = (unsigned)v, hi = (unsigned)(v >> 32);
  lo = __shfl_xor(lo, m, 32);
  hi = __shfl_xor(hi, m, 32);
  return ((unsigned long long)hi << 32) | lo;
}

// ---------- 1) normalize feats over channel dim ----------
__global__ void k_norm_feats(const float* __restrict__ feats, float* __restrict__ nf) {
  int i = blockIdx.x * blockDim.x + threadIdx.x;   // 2*4800 positions
  if (i >= 2 * SP) return;
  int b = i / SP, s = i % SP;
  const float* p = feats + (size_t)b * NCH * SP + s;
  float ss = 0.f;
  #pragma unroll 8
  for (int c = 0; c < NCH; ++c) { float v = p[c * SP]; ss += v * v; }
  float nrm = sqrtf(ss);
  float sc = 1.f / fmaxf(nrm, 1e-12f);
  float* q = nf + (size_t)b * NCH * SP + s;
  #pragma unroll 8
  for (int c = 0; c < NCH; ++c) q[c * SP] = p[c * SP] * sc;
}

// ---------- 2) softmax(65) -> take 64 -> pixel_shuffle(8) ----------
__global__ void k_softmax_shuffle(const float* __restrict__ logits, float* __restrict__ heat) {
  int i = blockIdx.x * blockDim.x + threadIdx.x;   // 2*4800 positions
  if (i >= 2 * SP) return;
  int b = i / SP, s = i % SP;
  int h = s / WF, w = s % WF;
  const float* p = logits + (size_t)b * 65 * SP + s;
  float mx = -3.4e38f;
  for (int c = 0; c < 65; ++c) mx = fmaxf(mx, p[c * SP]);
  float sum = 0.f;
  for (int c = 0; c < 65; ++c) sum += __expf(p[c * SP] - mx);
  float inv = 1.f / sum;
  float* hb = heat + (size_t)b * NPIX;
  for (int c = 0; c < NCH; ++c) {
    float v = __expf(p[c * SP] - mx) * inv;
    int oy = h * 8 + (c >> 3);
    int ox = w * 8 + (c & 7);
    hb[oy * WI + ox] = v;
  }
}

// ---------- 3) bilinear resize heatmap 60x80 -> 480x640 (half-pixel) ----------
__global__ void k_resize(const float* __restrict__ hm, float* __restrict__ out) {
  int i = blockIdx.x * blockDim.x + threadIdx.x;   // 2*307200
  if (i >= 2 * NPIX) return;
  int b = i / NPIX, r = i % NPIX;
  int oy = r / WI, ox = r % WI;
  float sy = fminf(fmaxf((oy + 0.5f) * 0.125f - 0.5f, 0.f), (float)(HF - 1));
  float sx = fminf(fmaxf((ox + 0.5f) * 0.125f - 0.5f, 0.f), (float)(WF - 1));
  int y0 = (int)floorf(sy), x0 = (int)floorf(sx);
  int y1 = min(y0 + 1, HF - 1), x1 = min(x0 + 1, WF - 1);
  float wy = sy - y0, wx = sx - x0;
  const float* p = hm + (size_t)b * SP;
  float v00 = p[y0 * WF + x0], v01 = p[y0 * WF + x1];
  float v10 = p[y1 * WF + x0], v11 = p[y1 * WF + x1];
  float v = v00 * (1 - wx) * (1 - wy) + v01 * wx * (1 - wy) + v10 * (1 - wx) * wy + v11 * wx * wy;
  out[(b ? O_R1 : O_R0) + r] = v;
}

// ---------- 4) 5x5 NMS on heat, multiply by reliability -> final scores ----------
__global__ void k_nms(const float* __restrict__ heat, float* __restrict__ out) {
  int i = blockIdx.x * blockDim.x + threadIdx.x;   // 2*307200
  if (i >= 2 * NPIX) return;
  int b = i / NPIX, r = i % NPIX;
  int y = r / WI, x = r % WI;
  const float* hb = heat + (size_t)b * NPIX;
  float c = hb[r];
  float lm = -3.4e38f;
  #pragma unroll
  for (int dy = -2; dy <= 2; ++dy) {
    int yy = y + dy;
    if (yy < 0 || yy >= HI) continue;
    #pragma unroll
    for (int dx = -2; dx <= 2; ++dx) {
      int xx = x + dx;
      if (xx < 0 || xx >= WI) continue;
      lm = fmaxf(lm, hb[yy * WI + xx]);
    }
  }
  float keep = (c == lm) ? c : 0.f;
  float rel = out[(b ? O_R1 : O_R0) + r];
  out[(b ? O_H1 : O_H0) + r] = keep * rel;
}

// ---------- 5) radix-select top-4096 ----------
__global__ void k_topk_init(Ctl* ctl, unsigned* hist, unsigned long long* sel) {
  int t = threadIdx.x;
  hist[t] = 0;
  for (int i = t; i < KSEL; i += 256) sel[i] = 0ull;
  if (t == 0) { ctl->prefix = 0ull; ctl->mask = 0ull; ctl->rank = KSEL; ctl->cnt = 0u; }
}

__global__ void k_hist(const float* __restrict__ sc, const Ctl* __restrict__ ctl,
                       unsigned* __restrict__ hist, int pass) {
  __shared__ unsigned sh[256];
  int t = threadIdx.x;
  sh[t] = 0;
  __syncthreads();
  int i = blockIdx.x * blockDim.x + t;
  int shift = 56 - 8 * pass;
  unsigned long long pre = ctl->prefix, msk = ctl->mask;
  if (i < NPIX) {
    unsigned long long key = ((unsigned long long)fordf(sc[i]) << 32) | (0xFFFFFFFFu - (unsigned)i);
    if ((key & msk) == pre) {
      unsigned d = (unsigned)((key >> shift) & 0xFFull);
      atomicAdd(&sh[d], 1u);
    }
  }
  __syncthreads();
  if (sh[t]) atomicAdd(&hist[t], sh[t]);
}

__global__ void k_scan(Ctl* ctl, unsigned* hist, int pass) {
  int shift = 56 - 8 * pass;
  unsigned rank = ctl->rank;
  unsigned long long above = 0;
  int selbin = 0;
  for (int bin = 255; bin >= 0; --bin) {
    unsigned c = hist[bin];
    if (above + c >= rank) { selbin = bin; break; }
    above += c;
  }
  ctl->rank = rank - (unsigned)above;
  ctl->prefix |= ((unsigned long long)selbin) << shift;
  ctl->mask |= (0xFFull << shift);
  for (int i = 0; i < 256; ++i) hist[i] = 0;
}

__global__ void k_compact(const float* __restrict__ sc, Ctl* ctl,
                          unsigned long long* __restrict__ sel) {
  int i = blockIdx.x * blockDim.x + threadIdx.x;
  if (i >= NPIX) return;
  unsigned long long key = ((unsigned long long)fordf(sc[i]) << 32) | (0xFFFFFFFFu - (unsigned)i);
  if (key >= ctl->prefix) {
    unsigned pos = atomicAdd(&ctl->cnt, 1u);
    if (pos < KSEL) sel[pos] = key;
  }
}

// descending bitonic sort of 4096 u64 keys, one workgroup
__global__ void __launch_bounds__(1024) k_bitonic(unsigned long long* sel) {
  __shared__ unsigned long long s[KSEL];
  int t = threadIdx.x;
  for (int e = 0; e < 4; ++e) s[t + e * 1024] = sel[t + e * 1024];
  __syncthreads();
  for (int k = 2; k <= KSEL; k <<= 1) {
    for (int j = k >> 1; j > 0; j >>= 1) {
      for (int e = 0; e < 4; ++e) {
        int i = t + e * 1024;
        int ixj = i ^ j;
        if (ixj > i) {
          unsigned long long a = s[i], b = s[ixj];
          bool descBlock = ((i & k) == 0);
          if ((a < b) == descBlock) { s[i] = b; s[ixj] = a; }
        }
      }
      __syncthreads();
    }
  }
  for (int e = 0; e < 4; ++e) sel[t + e * 1024] = s[t + e * 1024];
}

// ---------- 6) keypoints + grid-sampled L2-normalized f16 descriptors ----------
__global__ void k_desc(const unsigned long long* __restrict__ sel,
                       const float* __restrict__ nf,
                       _Float16* __restrict__ desc,
                       float* __restrict__ out) {
  int wid = (blockIdx.x * blockDim.x + threadIdx.x) >> 5;  // 8192 waves
  int lane = threadIdx.x & 31;
  int b = wid >> 12, m = wid & 4095;
  unsigned long long key = sel[(size_t)b * KSEL + m];
  unsigned idx = 0xFFFFFFFFu - (unsigned)(key & 0xFFFFFFFFull);
  int px = (int)(idx % WI), py = (int)(idx / WI);
  if (lane == 0) {
    out[(b ? O_KPTS1 : O_KPTS0) + m * 2 + 0] = (float)px;
    out[(b ? O_KPTS1 : O_KPTS0) + m * 2 + 1] = (float)py;
  }
  float ix = (float)px * ((float)(WF - 1) / (float)(WI - 1));
  float iy = (float)py * ((float)(HF - 1) / (float)(HI - 1));
  int x0 = (int)floorf(ix), y0 = (int)floorf(iy);
  float wx = ix - x0, wy = iy - y0;
  int x1 = min(x0 + 1, WF - 1), y1 = min(y0 + 1, HF - 1);
  float w00 = (1 - wx) * (1 - wy), w01 = wx * (1 - wy), w10 = (1 - wx) * wy, w11 = wx * wy;
  const float* base = nf + (size_t)b * NCH * SP;
  int c0 = lane, c1 = lane + 32;
  const float* p0 = base + (size_t)c0 * SP;
  const float* p1 = base + (size_t)c1 * SP;
  float v0 = p0[y0 * WF + x0] * w00 + p0[y0 * WF + x1] * w01 + p0[y1 * WF + x0] * w10 + p0[y1 * WF + x1] * w11;
  float v1 = p1[y0 * WF + x0] * w00 + p1[y0 * WF + x1] * w01 + p1[y1 * WF + x0] * w10 + p1[y1 * WF + x1] * w11;
  float ss = v0 * v0 + v1 * v1;
  for (int msk = 1; msk < 32; msk <<= 1) ss += __shfl_xor(ss, msk, 32);
  float sc = 1.f / fmaxf(sqrtf(ss), 1e-12f);
  _Float16* d = desc + ((size_t)b * KSEL + m) * NCH;
  d[c0] = (_Float16)(v0 * sc);
  d[c1] = (_Float16)(v1 * sc);
}

__global__ void k_init_colmax(unsigned long long* colmax) {
  int i = blockIdx.x * blockDim.x + threadIdx.x;
  if (i < KSEL) colmax[i] = 0ull;
}

// ---------- 7) fused WMMA GEMM (4096x4096x64) + row/col argmax ----------
union HV { v16h v; float4 q[2]; };
union CV { v8f v; float f[8]; };

__global__ void __launch_bounds__(256) k_sim(const _Float16* __restrict__ desc,
                                             unsigned long long* __restrict__ rowmax,
                                             unsigned long long* __restrict__ colmax) {
  __shared__ unsigned long long rb[8][16];
  const _Float16* d0 = desc;
  const _Float16* d1 = desc + (size_t)KSEL * NCH;
  int lane = threadIdx.x & 31;
  int wave = threadIdx.x >> 5;
  int r = lane & 15;
  int h8 = (lane >> 4) * 8;
  int rowbase = blockIdx.x * 16;

  // A tile regs: row (rowbase + r), K in WMMA 16x32 f16 layout
  const _Float16* pa = d0 + (size_t)(rowbase + r) * NCH;
  HV a0, a1;
  a0.q[0] = *(const float4*)(pa + h8);
  a0.q[1] = *(const float4*)(pa + 16 + h8);
  a1.q[0] = *(const float4*)(pa + 32 + h8);
  a1.q[1] = *(const float4*)(pa + 48 + h8);

  unsigned long long rowbest[8];
  #pragma unroll
  for (int j = 0; j < 8; ++j) rowbest[j] = 0ull;

  // prologue: load first B tile
  HV b0c, b1c;
  {
    const _Float16* pb = d1 + (size_t)(wave * 16 + r) * NCH;
    b0c.q[0] = *(const float4*)(pb + h8);
    b0c.q[1] = *(const float4*)(pb + 16 + h8);
    b1c.q[0] = *(const float4*)(pb + 32 + h8);
    b1c.q[1] = *(const float4*)(pb + 48 + h8);
  }

  for (int ct = wave; ct < KSEL / 16; ct += 8) {
    // software pipeline: issue next tile's loads before consuming current regs
    HV b0n = b0c, b1n = b1c;
    int ctn = ct + 8;
    if (ctn < KSEL / 16) {
      const _Float16* pn = d1 + (size_t)(ctn * 16 + r) * NCH;
      b0n.q[0] = *(const float4*)(pn + h8);
      b0n.q[1] = *(const float4*)(pn + 16 + h8);
      b1n.q[0] = *(const float4*)(pn + 32 + h8);
      b1n.q[1] = *(const float4*)(pn + 48 + h8);
    }

    v8f acc = {};
    acc = __builtin_amdgcn_wmma_f32_16x16x32_f16(false, a0.v, false, b0c.v, (short)0, acc, false, false);
    acc = __builtin_amdgcn_wmma_f32_16x16x32_f16(false, a1.v, false, b1c.v, (short)0, acc, false, false);
    CV cu; cu.v = acc;

    int ncol = ct * 16 + r;
    unsigned notcol = 0xFFFFFFFFu - (unsigned)ncol;
    unsigned long long cbest = 0ull;
    // no shuffles in-loop: rowbest keeps per-lane-column-slot running max, key carries col
    #pragma unroll
    for (int j = 0; j < 8; ++j) {
      unsigned o = fordf(cu.f[j]);
      unsigned long long rkey = ((unsigned long long)o << 32) | notcol;
      if (rkey > rowbest[j]) rowbest[j] = rkey;
      unsigned long long ckey = ((unsigned long long)o << 32) |
                                (0xFFFFFFFFu - (unsigned)(rowbase + j + h8));
      if (ckey > cbest) cbest = ckey;
    }
    unsigned long long oth = shflxor_u64(cbest, 16);
    if (oth > cbest) cbest = oth;
    if (lane < 16) atomicMax(&colmax[ncol], cbest);

    b0c = b0n; b1c = b1n;
  }

  // single row butterfly after the tile loop (lanes 0-15: row j, lanes 16-31: row j+8)
  #pragma unroll
  for (int j = 0; j < 8; ++j) {
    unsigned long long k = rowbest[j];
    unsigned long long t;
    t = shflxor_u64(k, 1); if (t > k) k = t;
    t = shflxor_u64(k, 2); if (t > k) k = t;
    t = shflxor_u64(k, 4); if (t > k) k = t;
    t = shflxor_u64(k, 8); if (t > k) k = t;
    rowbest[j] = k;
  }
  if (lane == 0) {
    #pragma unroll
    for (int j = 0; j < 8; ++j) rb[wave][j] = rowbest[j];
  }
  if (lane == 16) {
    #pragma unroll
    for (int j = 0; j < 8; ++j) rb[wave][j + 8] = rowbest[j];
  }
  __syncthreads();
  if (threadIdx.x < 16) {
    unsigned long long m = rb[0][threadIdx.x];
    #pragma unroll
    for (int w = 1; w < 8; ++w) if (rb[w][threadIdx.x] > m) m = rb[w][threadIdx.x];
    rowmax[rowbase + threadIdx.x] = m;
  }
}

// ---------- 8) mutual match + outputs ----------
__global__ void k_finalize(const unsigned long long* __restrict__ rowmax,
                           const unsigned long long* __restrict__ colmax,
                           float* __restrict__ out) {
  int k = blockIdx.x * blockDim.x + threadIdx.x;
  if (k >= KSEL) return;
  unsigned long long rk = rowmax[k];
  float s0 = finvord((unsigned)(rk >> 32));
  unsigned m01 = 0xFFFFFFFFu - (unsigned)(rk & 0xFFFFFFFFull);
  unsigned long long ck = colmax[m01 & 4095];
  unsigned m10 = 0xFFFFFFFFu - (unsigned)(ck & 0xFFFFFFFFull);
  bool valid = (m10 == (unsigned)k) && (s0 > 0.1f);
  out[O_MATCH01 + k] = (float)m01;
  out[O_SCORE0 + k] = s0;
  out[O_VALID + k] = valid ? 1.f : 0.f;
}

extern "C" void kernel_launch(void* const* d_in, const int* in_sizes, int n_in,
                              void* d_out, int out_size, void* d_ws, size_t ws_size,
                              hipStream_t stream) {
  (void)in_sizes; (void)n_in; (void)out_size; (void)ws_size;
  const float* feats  = (const float*)d_in[0];
  const float* logits = (const float*)d_in[1];
  const float* hm     = (const float*)d_in[2];
  float* out = (float*)d_out;
  char* ws = (char*)d_ws;

  float* nf   = (float*)(ws + WS_NFEATS);
  float* heat = (float*)(ws + WS_HEAT);
  unsigned long long* sel    = (unsigned long long*)(ws + WS_SEL);
  _Float16* desc             = (_Float16*)(ws + WS_DESC);
  unsigned long long* rowmax = (unsigned long long*)(ws + WS_ROWMAX);
  unsigned long long* colmax = (unsigned long long*)(ws + WS_COLMAX);
  unsigned* hist = (unsigned*)(ws + WS_HIST);
  Ctl* ctl = (Ctl*)(ws + WS_CTL);

  k_norm_feats<<<(2 * SP + 255) / 256, 256, 0, stream>>>(feats, nf);
  k_softmax_shuffle<<<(2 * SP + 255) / 256, 256, 0, stream>>>(logits, heat);
  k_resize<<<(2 * NPIX + 255) / 256, 256, 0, stream>>>(hm, out);
  k_nms<<<(2 * NPIX + 255) / 256, 256, 0, stream>>>(heat, out);

  for (int b = 0; b < 2; ++b) {
    const float* sc = out + (b ? O_H1 : O_H0);
    unsigned long long* selb = sel + (size_t)b * KSEL;
    k_topk_init<<<1, 256, 0, stream>>>(ctl, hist, selb);
    for (int p = 0; p < 8; ++p) {
      k_hist<<<NPIX / 256, 256, 0, stream>>>(sc, ctl, hist, p);
      k_scan<<<1, 1, 0, stream>>>(ctl, hist, p);
    }
    k_compact<<<NPIX / 256, 256, 0, stream>>>(sc, ctl, selb);
    k_bitonic<<<1, 1024, 0, stream>>>(selb);
  }

  k_desc<<<1024, 256, 0, stream>>>(sel, nf, desc, out);
  k_init_colmax<<<16, 256, 0, stream>>>(colmax);
  k_sim<<<KSEL / 16, 256, 0, stream>>>(desc, rowmax, colmax);
  k_finalize<<<16, 256, 0, stream>>>(rowmax, colmax, out);
}
